// GCNRegression_74165495267797
// MI455X (gfx1250) — matrix operand order repaired and checked by reading, not compiled, not run on hardware
//
#include <hip/hip_runtime.h>
#include <hip/hip_bf16.h>

typedef __attribute__((ext_vector_type(16))) __bf16 v16bf;
typedef __attribute__((ext_vector_type(8)))  __bf16 v8bf;
typedef __attribute__((ext_vector_type(8)))  float  v8f;

// ---------------- utility kernels ----------------

__global__ void fill_f32_kernel(float* __restrict__ p, float v, long long n) {
    long long i = (long long)blockIdx.x * blockDim.x + threadIdx.x;
    if (i < n) p[i] = v;
}

__global__ void deg_kernel(const int* __restrict__ row, float* __restrict__ deg, int E) {
    int e = blockIdx.x * blockDim.x + threadIdx.x;
    if (e < E) atomicAdd(&deg[row[e]], 1.0f);
}

__global__ void norm_kernel(const int* __restrict__ row, const int* __restrict__ col,
                            const float* __restrict__ deg, float* __restrict__ norm, int E) {
    int e = blockIdx.x * blockDim.x + threadIdx.x;
    if (e < E) {
        // deg**-0.5: 0-degree -> +inf, matching reference semantics
        norm[e] = rsqrtf(deg[row[e]]) * rsqrtf(deg[col[e]]);
    }
}

// fp32 -> bf16 with round-to-nearest-even (manual, avoids relying on cast lowering)
__global__ void f32_to_bf16_kernel(const float* __restrict__ in,
                                   unsigned short* __restrict__ out, long long n) {
    long long i = (long long)blockIdx.x * blockDim.x + threadIdx.x;
    if (i < n) {
        unsigned int u = __float_as_uint(in[i]);
        u += 0x7FFFu + ((u >> 16) & 1u);
        out[i] = (unsigned short)(u >> 16);
    }
}

// ---------------- WMMA GEMM:  out[N,Out] = Xbf16[N,K] @ Wbf16[Out,K]^T + bias ----------------
// One 16x16 output tile per wave; K consumed 32 at a time with v_wmma_f32_16x16x32_bf16.
// Fragment layouts used here (A and B share one contiguous k order; C/D as in the ISA):
//   A 16x32 bf16 : lane<16 -> M=lane, K = h*16 .. h*16+15 (contiguous)  (h = lane>>4)
//   B 32x16 bf16 : lane<16 -> N=lane, K = h*16 .. h*16+15 (contiguous)
//   C/D 16x16 f32: VGPR r -> M = r + h*8, N = lane&15
__global__ __launch_bounds__(128)
void gemm_bf16_wmma_kernel(const unsigned short* __restrict__ xb,   // [Nrows,K] bf16 bits
                           const unsigned short* __restrict__ wb,   // [Out,K]   bf16 bits
                           const float* __restrict__ bias,          // [Out]
                           float* __restrict__ out,                 // [Nrows,Out]
                           int K, int OutDim, int Mtiles) {
    const int wave  = threadIdx.x >> 5;
    const int lane  = threadIdx.x & 31;
    const int tileM = blockIdx.x * 4 + wave;
    if (tileM >= Mtiles) return;              // wave-uniform: EXEC stays all-ones for WMMA
    const int tileN = blockIdx.y;
    const int half  = lane >> 4;
    const int l16   = lane & 15;

    const int rowA = tileM * 16 + l16;        // A-fragment row (M = lane&15)
    const int colB = tileN * 16 + l16;        // B-fragment / output column (N = lane&15)

    const unsigned short* aBase = xb + (size_t)rowA * K + half * 16;
    const unsigned short* bBase = wb + (size_t)colB * K + half * 16;

    const float bv = bias[colB];
    v8f c = {bv, bv, bv, bv, bv, bv, bv, bv};

    #pragma unroll 8
    for (int k0 = 0; k0 < K; k0 += 32) {
        v8bf  a0 = *(const v8bf*)(aBase + k0);        // K = k0 + h*16 .. +7       (16B)
        v8bf  a1 = *(const v8bf*)(aBase + k0 + 8);    // K = k0 + h*16 + 8 .. +7   (16B)
        v16bf a  = __builtin_shufflevector(a0, a1, 0,1,2,3,4,5,6,7,8,9,10,11,12,13,14,15);
        v16bf b  = *(const v16bf*)(bBase + k0);       // K = k0 + h*16 .. +15      (32B)
        c = __builtin_amdgcn_wmma_f32_16x16x32_bf16(
                /*neg_a=*/false, a, /*neg_b=*/false, b,
                /*c_mod=*/(short)0, c, /*reuse_a=*/false, /*reuse_b=*/false);
    }

    float* o = out + (size_t)(tileM * 16 + half * 8) * OutDim + colB;
    #pragma unroll
    for (int r = 0; r < 8; ++r)
        o[(size_t)r * OutDim] = c[r];                 // half-wave-coalesced f32 stores
}

// ---------------- edge gather/scale/scatter:  hout[col] += norm * hlin[row] ----------------
__global__ void aggregate_kernel(const float* __restrict__ hlin, const float* __restrict__ norm,
                                 const int* __restrict__ row, const int* __restrict__ col,
                                 float* __restrict__ hout, int E, int OutDim) {
    const int chunks = OutDim >> 2;                  // float4 chunks per row
    long long idx = (long long)blockIdx.x * blockDim.x + threadIdx.x;
    long long total = (long long)E * chunks;
    if (idx >= total) return;
    int e  = (int)(idx / chunks);
    int c4 = (int)(idx % chunks) << 2;               // consecutive threads -> coalesced row read
    float nv = norm[e];
    int r  = row[e];
    int cl = col[e];
    float4 v = *(const float4*)(hlin + (size_t)r * OutDim + c4);
    float* dst = hout + (size_t)cl * OutDim + c4;
    atomicAdd(dst + 0, nv * v.x);
    atomicAdd(dst + 1, nv * v.y);
    atomicAdd(dst + 2, nv * v.z);
    atomicAdd(dst + 3, nv * v.w);
}

// ---------------- graph pooling ----------------

__global__ void pool_kernel(const float* __restrict__ h, const int* __restrict__ batch,
                            float* __restrict__ pool, int N, int OutDim) {
    const int chunks = OutDim >> 2;
    long long idx = (long long)blockIdx.x * blockDim.x + threadIdx.x;
    long long total = (long long)N * chunks;
    if (idx >= total) return;
    int n  = (int)(idx / chunks);
    int c4 = (int)(idx % chunks) << 2;
    int g  = batch[n];
    float4 v = *(const float4*)(h + (size_t)n * OutDim + c4);
    float* dst = pool + (size_t)g * OutDim + c4;
    atomicAdd(dst + 0, v.x);
    atomicAdd(dst + 1, v.y);
    atomicAdd(dst + 2, v.z);
    atomicAdd(dst + 3, v.w);
}

__global__ void counts_kernel(const int* __restrict__ batch, float* __restrict__ counts, int N) {
    int n = blockIdx.x * blockDim.x + threadIdx.x;
    if (n < N) atomicAdd(&counts[batch[n]], 1.0f);
}

// one block (128 threads) per graph: out[g] = dot(pool[g]/maxCount, lin_w) + lin_b
__global__ void final_kernel(const float* __restrict__ pool, const float* __restrict__ counts,
                             const float* __restrict__ lin_w, const float* __restrict__ lin_b,
                             float* __restrict__ out, int G, int OutDim) {
    __shared__ float s[128];
    const int g = blockIdx.x;
    const int t = threadIdx.x;

    float mx = 0.0f;
    for (int i = t; i < G; i += blockDim.x) mx = fmaxf(mx, counts[i]);
    s[t] = mx;
    __syncthreads();
    for (int st = blockDim.x >> 1; st > 0; st >>= 1) {
        if (t < st) s[t] = fmaxf(s[t], s[t + st]);
        __syncthreads();
    }
    mx = s[0];
    __syncthreads();

    float v = (t < OutDim) ? (pool[(size_t)g * OutDim + t] / mx) * lin_w[t] : 0.0f;
    s[t] = v;
    __syncthreads();
    for (int st = blockDim.x >> 1; st > 0; st >>= 1) {
        if (t < st) s[t] = s[t] + s[t + st];
        __syncthreads();
    }
    if (t == 0) out[g] = s[0] + lin_b[0];
}

// ---------------- host orchestration ----------------

static inline int cdiv(long long a, long long b) { return (int)((a + b - 1) / b); }

extern "C" void kernel_launch(void* const* d_in, const int* in_sizes, int n_in,
                              void* d_out, int out_size, void* d_ws, size_t ws_size,
                              hipStream_t stream) {
    const float* x      = (const float*)d_in[0];
    const int*   edges  = (const int*)d_in[1];
    const int*   batch  = (const int*)d_in[2];
    const float* W[4]   = {(const float*)d_in[4], (const float*)d_in[6],
                           (const float*)d_in[8], (const float*)d_in[10]};
    const float* B[4]   = {(const float*)d_in[5], (const float*)d_in[7],
                           (const float*)d_in[9], (const float*)d_in[11]};
    const float* lin_w  = (const float*)d_in[12];
    const float* lin_b  = (const float*)d_in[13];
    float*       out    = (float*)d_out;

    const int N   = in_sizes[2];          // batch has one entry per node
    const int E   = in_sizes[1] / 2;      // edge_index is [2,E]
    const int IN  = (int)((long long)in_sizes[0] / N);
    const int HID = in_sizes[5];          // b1
    const int OUT = in_sizes[11];         // b4
    const int G   = out_size;             // [G,1] output

    const int* row = edges;
    const int* col = edges + E;

    // workspace carve-up (256B aligned)
    char* ws = (char*)d_ws;
    size_t off = 0;
    auto carve = [&](size_t bytes) -> char* {
        char* p = ws + off;
        off = (off + bytes + 255) & ~(size_t)255;
        return p;
    };
    const int maxK = (IN > HID) ? IN : HID;
    float*          deg    = (float*)carve((size_t)N * 4);
    float*          norm   = (float*)carve((size_t)E * 4);
    unsigned short* xb     = (unsigned short*)carve((size_t)N * maxK * 2);
    unsigned short* wb     = (unsigned short*)carve((size_t)HID * maxK * 2);
    float*          hlin   = (float*)carve((size_t)N * HID * 4);
    float*          hA     = (float*)carve((size_t)N * HID * 4);
    float*          hB     = (float*)carve((size_t)N * HID * 4);
    float*          pool   = (float*)carve((size_t)G * OUT * 4);
    float*          counts = (float*)carve((size_t)G * 4);
    (void)ws_size; (void)n_in;

    const int TB = 256;

    // 1) degrees + per-edge norm
    fill_f32_kernel<<<cdiv(N, TB), TB, 0, stream>>>(deg, 0.0f, N);
    deg_kernel<<<cdiv(E, TB), TB, 0, stream>>>(row, deg, E);
    norm_kernel<<<cdiv(E, TB), TB, 0, stream>>>(row, col, deg, norm, E);

    // 2) four GCN layers
    const float* cur = x;
    float* bufs[2] = {hA, hB};
    for (int l = 0; l < 4; ++l) {
        const int K      = (l == 0) ? IN : HID;
        const int OutDim = (l == 3) ? OUT : HID;
        float* hout = bufs[l & 1];

        long long nx = (long long)N * K;
        long long nw = (long long)OutDim * K;
        f32_to_bf16_kernel<<<cdiv(nx, TB), TB, 0, stream>>>(cur, xb, nx);
        f32_to_bf16_kernel<<<cdiv(nw, TB), TB, 0, stream>>>(W[l], wb, nw);

        const int Mtiles = cdiv(N, 16);                 // N is a multiple of 16
        dim3 gg(cdiv(Mtiles, 4), OutDim / 16);
        gemm_bf16_wmma_kernel<<<gg, 128, 0, stream>>>(xb, wb, B[l], hlin, K, OutDim, Mtiles);

        long long nh = (long long)N * OutDim;
        fill_f32_kernel<<<cdiv(nh, TB), TB, 0, stream>>>(hout, 0.0f, nh);
        long long na = (long long)E * (OutDim >> 2);
        aggregate_kernel<<<cdiv(na, TB), TB, 0, stream>>>(hlin, norm, row, col, hout, E, OutDim);

        cur = hout;
    }

    // 3) graph pooling + final linear
    fill_f32_kernel<<<cdiv((long long)G * OUT, TB), TB, 0, stream>>>(pool, 0.0f, (long long)G * OUT);
    fill_f32_kernel<<<cdiv(G, TB), TB, 0, stream>>>(counts, 0.0f, G);
    long long np = (long long)N * (OUT >> 2);
    pool_kernel<<<cdiv(np, TB), TB, 0, stream>>>(cur, batch, pool, N, OUT);
    counts_kernel<<<cdiv(N, TB), TB, 0, stream>>>(batch, counts, N);
    final_kernel<<<G, 128, 0, stream>>>(pool, counts, lin_w, lin_b, out, G, OUT);
}
